// NCEAverage2_36026185679484
// MI455X (gfx1250) — matrix operand order repaired and checked by reading, not compiled, not run on hardware
//
#include <hip/hip_runtime.h>
#include <math.h>

typedef __attribute__((ext_vector_type(16))) _Float16     v16h;
typedef __attribute__((ext_vector_type(8)))  float        v8f;
typedef __attribute__((ext_vector_type(8)))  unsigned int v8u;

#define B_SZ   256
#define D_SZ   128
#define KP1    4097      // K+1
#define NROWS  200000
#define NTILES 257       // ceil(4097/16)
#define TPW    8         // k-tiles per wave
#define NGRP   33        // ceil(NTILES/TPW)

// Two fp32 words -> two packed f16 values in {-1.0, +1.0} selected by the
// fp32 sign bits. (Bank rows are exactly +-1/sqrt(128): sign() never sees 0.)
__device__ __forceinline__ unsigned int sgnpack(unsigned int u0, unsigned int u1) {
  return 0x3C003C00u | ((u0 >> 16) & 0x8000u) | (u1 & 0x80000000u);
}

__device__ __forceinline__ void put_cvt4(v16h& a, int t0, float4 x) {
  a[t0 + 0] = (_Float16)x.x; a[t0 + 1] = (_Float16)x.y;
  a[t0 + 2] = (_Float16)x.z; a[t0 + 3] = (_Float16)x.w;
}

__device__ __forceinline__ v16h frag_from(uint4 lo, uint4 hi) {
  v8u w;
  w[0] = lo.x; w[1] = lo.y; w[2] = lo.z; w[3] = lo.w;
  w[4] = hi.x; w[5] = hi.y; w[6] = hi.z; w[7] = hi.w;
  return __builtin_bit_cast(v16h, w);
}

// ---------------------------------------------------------------------------
// Fused bank copy + binarize: stream the fp32 banks into the output region
// (required by the reference) and simultaneously emit f16 +-1 sign banks into
// workspace. Adds only the sign-write traffic on top of the mandatory copy.
// ---------------------------------------------------------------------------
__global__ __launch_bounds__(256)
void copy_binarize_kernel(const uint4* __restrict__ srcL, const uint4* __restrict__ srcA,
                          uint4* __restrict__ dstL, uint4* __restrict__ dstA,
                          uint4* __restrict__ sgnL, uint4* __restrict__ sgnA, long n8) {
  long i = (long)blockIdx.x * blockDim.x + threadIdx.x;
  const long stride = (long)gridDim.x * blockDim.x;
  for (; i < n8; i += stride) {
    const uint4 a0 = srcL[2 * i], a1 = srcL[2 * i + 1];
    dstL[2 * i] = a0; dstL[2 * i + 1] = a1;
    sgnL[i] = make_uint4(sgnpack(a0.x, a0.y), sgnpack(a0.z, a0.w),
                         sgnpack(a1.x, a1.y), sgnpack(a1.z, a1.w));
    const uint4 b0 = srcA[2 * i], b1 = srcA[2 * i + 1];
    dstA[2 * i] = b0; dstA[2 * i + 1] = b1;
    sgnA[i] = make_uint4(sgnpack(b0.x, b0.y), sgnpack(b0.z, b0.w),
                         sgnpack(b1.x, b1.y), sgnpack(b1.z, b1.w));
  }
}

// ---------------------------------------------------------------------------
// Fast path: one wave32 per (b, group of 8 k-tiles). A-fragments are straight
// uint4 loads from the pre-binarized f16 banks (no per-element VALU); B = the
// f16 sample vector replicated across all 16 columns, held in registers for
// the whole group. All 16 row loads of a tile are staged before the 8 WMMAs
// so the wave keeps many loads in flight.
// ---------------------------------------------------------------------------
__global__ __launch_bounds__(256)
void nce_scores_f16_kernel(const float* __restrict__ l, const float* __restrict__ ab,
                           const int* __restrict__ idx,
                           const unsigned short* __restrict__ sgnL,
                           const unsigned short* __restrict__ sgnA,
                           float* __restrict__ outLAB, float* __restrict__ outABL,
                           float invT) {
  const int wave = blockIdx.x * (blockDim.x >> 5) + (threadIdx.x >> 5);
  const int lane = threadIdx.x & 31;
  const int b    = wave / NGRP;             // uniform per wave
  const int grp  = wave - b * NGRP;
  if (b >= B_SZ) return;                    // wave-uniform; EXEC all-1s below

  const int m = lane & 15;
  const int g = lane >> 4;

  // Hoisted B operands: f16(ab[b]) / f16(l[b]), K=8g..8g+7 then 16+8g..16+8g+7
  // per chunk, identical in every column.
  v16h bAB[4], bL[4];
#pragma unroll
  for (int c = 0; c < 4; ++c) {
    const float4* p = (const float4*)(ab + (size_t)b * D_SZ + c * 32 + 8 * g);
    const float4* q = (const float4*)(l  + (size_t)b * D_SZ + c * 32 + 8 * g);
    put_cvt4(bAB[c], 0, p[0]); put_cvt4(bAB[c], 4, p[1]);
    put_cvt4(bAB[c], 8, p[4]); put_cvt4(bAB[c], 12, p[5]);
    put_cvt4(bL[c], 0, q[0]); put_cvt4(bL[c], 4, q[1]);
    put_cvt4(bL[c], 8, q[4]); put_cvt4(bL[c], 12, q[5]);
  }

  // Preload all row indices for the group (keeps idx latency off the
  // per-tile critical path).
  const int* idxb = idx + (size_t)b * KP1;
  int rows[TPW];
#pragma unroll
  for (int t = 0; t < TPW; ++t) {
    int kk = (grp * TPW + t) * 16 + m;
    if (kk > KP1 - 1) kk = KP1 - 1;         // clamp tail loads
    rows[t] = idxb[kk];
  }

#pragma unroll 1
  for (int t = 0; t < TPW; ++t) {
    const int k0 = (grp * TPW + t) * 16;
    const unsigned short* rl = sgnL + (size_t)rows[t] * D_SZ;
    const unsigned short* ra = sgnA + (size_t)rows[t] * D_SZ;

    // Stage all 16 b128 loads of this tile (one clause, max MLP).
    // A layout (16-bit 16x32): halfs 0..7 -> K=8g..8g+7,
    //                          halfs 8..15 -> K=16+8g..16+8g+7
    uint4 dl[8], da[8];
#pragma unroll
    for (int c = 0; c < 4; ++c) {
      dl[2 * c]     = *(const uint4*)(rl + c * 32 + 8 * g);
      dl[2 * c + 1] = *(const uint4*)(rl + c * 32 + 16 + 8 * g);
      da[2 * c]     = *(const uint4*)(ra + c * 32 + 8 * g);
      da[2 * c + 1] = *(const uint4*)(ra + c * 32 + 16 + 8 * g);
    }

    v8f accL = {};   // weight_l . ab
    v8f accA = {};   // weight_ab . l
#pragma unroll
    for (int c = 0; c < 4; ++c) {
      accL = __builtin_amdgcn_wmma_f32_16x16x32_f16(
          false, frag_from(dl[2 * c], dl[2 * c + 1]), false, bAB[c],
          (short)0, accL, false, false);
      accA = __builtin_amdgcn_wmma_f32_16x16x32_f16(
          false, frag_from(da[2 * c], da[2 * c + 1]), false, bL[c],
          (short)0, accA, false, false);
    }

    // D layout: VGPR r holds M=r+8g, column N=lane&15; column 0 (lanes 0/16)
    // carries the 16 dot products (all columns identical by construction).
    float* pL = outLAB + (size_t)b * KP1 + k0 + 8 * g;
    float* pA = outABL + (size_t)b * KP1 + k0 + 8 * g;
    if (k0 + 16 <= KP1) {        // wave-uniform: branch-free store path
      if (m == 0) {
#pragma unroll
        for (int r = 0; r < 8; ++r) {
          pL[r] = accL[r] * invT;
          pA[r] = accA[r] * invT;
        }
      }
    } else {                     // only the very last tile
      if (m == 0) {
#pragma unroll
        for (int r = 0; r < 8; ++r) {
          if (k0 + 8 * g + r < KP1) {
            pL[r] = accL[r] * invT;
            pA[r] = accA[r] * invT;
          }
        }
      }
    }
  }
}

// ---------------------------------------------------------------------------
// Fallback path (workspace too small): binarize on the fly from fp32 banks.
// ---------------------------------------------------------------------------
__device__ __forceinline__ v16h sgn16(uint4 x0, uint4 x1, uint4 x2, uint4 x3) {
  v8u w;
  w[0] = sgnpack(x0.x, x0.y); w[1] = sgnpack(x0.z, x0.w);
  w[2] = sgnpack(x1.x, x1.y); w[3] = sgnpack(x1.z, x1.w);
  w[4] = sgnpack(x2.x, x2.y); w[5] = sgnpack(x2.z, x2.w);
  w[6] = sgnpack(x3.x, x3.y); w[7] = sgnpack(x3.z, x3.w);
  return __builtin_bit_cast(v16h, w);
}

__global__ __launch_bounds__(256)
void nce_scores_f32_kernel(const float* __restrict__ l, const float* __restrict__ ab,
                           const int* __restrict__ idx,
                           const float* __restrict__ memL, const float* __restrict__ memA,
                           float* __restrict__ outLAB, float* __restrict__ outABL,
                           float invT) {
  const int wave = blockIdx.x * (blockDim.x >> 5) + (threadIdx.x >> 5);
  const int lane = threadIdx.x & 31;
  const int b    = wave / NGRP;
  const int grp  = wave - b * NGRP;
  if (b >= B_SZ) return;

  const int m = lane & 15;
  const int g = lane >> 4;

  v16h bAB[4], bL[4];
#pragma unroll
  for (int c = 0; c < 4; ++c) {
    const float4* p = (const float4*)(ab + (size_t)b * D_SZ + c * 32 + 8 * g);
    const float4* q = (const float4*)(l  + (size_t)b * D_SZ + c * 32 + 8 * g);
    put_cvt4(bAB[c], 0, p[0]); put_cvt4(bAB[c], 4, p[1]);
    put_cvt4(bAB[c], 8, p[4]); put_cvt4(bAB[c], 12, p[5]);
    put_cvt4(bL[c], 0, q[0]); put_cvt4(bL[c], 4, q[1]);
    put_cvt4(bL[c], 8, q[4]); put_cvt4(bL[c], 12, q[5]);
  }

  const int* idxb = idx + (size_t)b * KP1;
  int rows[TPW];
#pragma unroll
  for (int t = 0; t < TPW; ++t) {
    int kk = (grp * TPW + t) * 16 + m;
    if (kk > KP1 - 1) kk = KP1 - 1;
    rows[t] = idxb[kk];
  }

#pragma unroll 1
  for (int t = 0; t < TPW; ++t) {
    const int k0 = (grp * TPW + t) * 16;
    const unsigned int* rl = (const unsigned int*)(memL + (size_t)rows[t] * D_SZ);
    const unsigned int* ra = (const unsigned int*)(memA + (size_t)rows[t] * D_SZ);

    v8f accL = {}, accA = {};
#pragma unroll
    for (int c = 0; c < 4; ++c) {
      const int d0 = c * 32;
      const uint4 lx0 = *(const uint4*)(rl + d0 + 8 * g);
      const uint4 lx1 = *(const uint4*)(rl + d0 + 8 * g + 4);
      const uint4 lx2 = *(const uint4*)(rl + d0 + 16 + 8 * g);
      const uint4 lx3 = *(const uint4*)(rl + d0 + 16 + 8 * g + 4);
      const uint4 ax0 = *(const uint4*)(ra + d0 + 8 * g);
      const uint4 ax1 = *(const uint4*)(ra + d0 + 8 * g + 4);
      const uint4 ax2 = *(const uint4*)(ra + d0 + 16 + 8 * g);
      const uint4 ax3 = *(const uint4*)(ra + d0 + 16 + 8 * g + 4);

      accL = __builtin_amdgcn_wmma_f32_16x16x32_f16(
          false, sgn16(lx0, lx1, lx2, lx3), false, bAB[c], (short)0, accL, false, false);
      accA = __builtin_amdgcn_wmma_f32_16x16x32_f16(
          false, sgn16(ax0, ax1, ax2, ax3), false, bL[c], (short)0, accA, false, false);
    }

    float* pL = outLAB + (size_t)b * KP1 + k0 + 8 * g;
    float* pA = outABL + (size_t)b * KP1 + k0 + 8 * g;
    if (k0 + 16 <= KP1) {
      if (m == 0) {
#pragma unroll
        for (int r = 0; r < 8; ++r) {
          pL[r] = accL[r] * invT;
          pA[r] = accA[r] * invT;
        }
      }
    } else {
      if (m == 0) {
#pragma unroll
        for (int r = 0; r < 8; ++r) {
          if (k0 + 8 * g + r < KP1) {
            pL[r] = accL[r] * invT;
            pA[r] = accA[r] * invT;
          }
        }
      }
    }
  }
}

// Plain copy (fallback path only).
__global__ __launch_bounds__(256)
void copy_mem_kernel(const float4* __restrict__ srcL, const float4* __restrict__ srcA,
                     float4* __restrict__ dstL, float4* __restrict__ dstA, long n4) {
  long i = (long)blockIdx.x * blockDim.x + threadIdx.x;
  const long stride = (long)gridDim.x * blockDim.x;
  for (; i < n4; i += stride) {
    dstL[i] = srcL[i];
    dstA[i] = srcA[i];
  }
}

__device__ __forceinline__ float block_sum_128(float v, float* red) {
  const int d = threadIdx.x;
  red[d] = v;
  __syncthreads();
#pragma unroll
  for (int s = 64; s > 0; s >>= 1) {
    if (d < s) red[d] += red[d + s];
    __syncthreads();
  }
  float r = red[0];
  __syncthreads();
  return r;
}

// One 128-thread block per sample b: momentum blend + renormalize row y[b].
__global__ __launch_bounds__(128)
void update_mem_kernel(const float* __restrict__ l, const float* __restrict__ ab,
                       const int* __restrict__ y,
                       const float* __restrict__ memL, const float* __restrict__ memA,
                       float* __restrict__ newL, float* __restrict__ newA) {
  __shared__ float red[128];
  const int b = blockIdx.x;
  const int d = threadIdx.x;
  const int row = y[b];

  const float lvv = l[(size_t)b * D_SZ + d];
  const float avv = ab[(size_t)b * D_SZ + d];

  const float nl = sqrtf(block_sum_128(lvv * lvv, red));
  const float na = sqrtf(block_sum_128(avv * avv, red));
  const float ln = lvv / nl;
  const float an = avv / na;

  const float pl = memL[(size_t)row * D_SZ + d] * 0.5f + ln * 0.5f;
  const float pa = memA[(size_t)row * D_SZ + d] * 0.5f + an * 0.5f;

  const float npl = sqrtf(block_sum_128(pl * pl, red));
  const float npa = sqrtf(block_sum_128(pa * pa, red));

  newL[(size_t)row * D_SZ + d] = pl / npl;
  newA[(size_t)row * D_SZ + d] = pa / npa;
}

extern "C" void kernel_launch(void* const* d_in, const int* in_sizes, int n_in,
                              void* d_out, int out_size, void* d_ws, size_t ws_size,
                              hipStream_t stream) {
  const float* l    = (const float*)d_in[0];
  const float* ab   = (const float*)d_in[1];
  const int*   y    = (const int*)d_in[2];
  const int*   idx  = (const int*)d_in[3];
  const float* memL = (const float*)d_in[4];
  const float* memA = (const float*)d_in[5];

  float* out = (float*)d_out;
  const long S1 = (long)B_SZ * KP1;                 // 1,048,832
  float* outLAB = out;
  float* outABL = out + S1;
  float* newL   = out + 2 * S1;
  float* newA   = out + 2 * S1 + (long)NROWS * D_SZ;

  const float invT = 1.0f / (0.07f * sqrtf((float)D_SZ));

  const int waves  = B_SZ * NGRP;                   // 8,448
  const int blocks = waves / 8;                     // 1,056 blocks of 256

  const size_t bankElems = (size_t)NROWS * D_SZ;    // 25,600,000
  const size_t sgnBytes  = bankElems * sizeof(unsigned short);  // 51.2 MB

  if (ws_size >= 2 * sgnBytes) {
    // Fast path: fused copy+binarize, then pure load+WMMA gather from the
    // L2-resident f16 sign banks (102.4 MB total < 192 MB L2).
    unsigned short* sgnL = (unsigned short*)d_ws;
    unsigned short* sgnA = (unsigned short*)((char*)d_ws + sgnBytes);
    const long n8 = (long)bankElems / 8;            // 3,200,000 (8 elems/iter)
    copy_binarize_kernel<<<12500, 256, 0, stream>>>(
        (const uint4*)memL, (const uint4*)memA,
        (uint4*)newL, (uint4*)newA, (uint4*)sgnL, (uint4*)sgnA, n8);
    nce_scores_f16_kernel<<<blocks, 256, 0, stream>>>(l, ab, idx, sgnL, sgnA,
                                                      outLAB, outABL, invT);
  } else {
    // Fallback: on-the-fly binarization from the fp32 banks.
    const long n4 = (long)bankElems / 4;            // 6,400,000 float4 per bank
    copy_mem_kernel<<<25000, 256, 0, stream>>>((const float4*)memL, (const float4*)memA,
                                               (float4*)newL, (float4*)newA, n4);
    nce_scores_f32_kernel<<<blocks, 256, 0, stream>>>(l, ab, idx, memL, memA,
                                                      outLAB, outABL, invT);
  }

  // Overwrite the 256 updated rows (ordered after the copy on `stream`).
  update_mem_kernel<<<B_SZ, 128, 0, stream>>>(l, ab, y, memL, memA, newL, newA);
}
